// GlobalAttentionBlock_35562329211765
// MI455X (gfx1250) — matrix operand order repaired and checked by reading, not compiled, not run on hardware
//
#include <hip/hip_runtime.h>
#include <hip/hip_bf16.h>
#include <stdint.h>

// ---------------------------------------------------------------------------
// Problem constants: B=16, C=384, H=W=64, heads=8, hd=48, window=8 (S=64),
// groups=4 (96 ch/group), EPS=1e-5.
// ---------------------------------------------------------------------------
#define TPB 256

typedef _Float16 hv8  __attribute__((ext_vector_type(8)));
typedef _Float16 hv16 __attribute__((ext_vector_type(16)));
typedef float    fv8  __attribute__((ext_vector_type(8)));
typedef uint32_t uv4  __attribute__((ext_vector_type(4)));
typedef uint32_t uv8  __attribute__((ext_vector_type(8)));

__device__ __forceinline__ hv16 cat16(hv8 lo, hv8 hi) {
  return __builtin_shufflevector(lo, hi, 0,1,2,3,4,5,6,7,8,9,10,11,12,13,14,15);
}

// A fragment from a lane-adjusted base (base already includes
// (lane&15)*stride + ((lane>>4)<<4)); off = K-slice offset in halves.
// Lane holds 16 contiguous K values K={kb..kb+15} (matches B).
__device__ __forceinline__ hv16 lda_off(const _Float16* p, int off) {
  hv8 lo = *(const hv8*)(p + off);
  hv8 hi = *(const hv8*)(p + off + 8);
  return cat16(lo, hi);
}
// B fragment from a lane-adjusted base (base includes (lane&15)*stride +
// ((lane>>4)<<4)); as in lda_off, lane holds 16 contiguous K values (this code's order).
__device__ __forceinline__ hv16 ldb_off(const _Float16* p, int off) {
  hv8 lo = *(const hv8*)(p + off);
  hv8 hi = *(const hv8*)(p + off + 8);
  return cat16(lo, hi);
}

#define WMMA_F16(A, Bm, Cacc) \
  __builtin_amdgcn_wmma_f32_16x16x32_f16(false, (A), false, (Bm), (short)0, (Cacc), false, false)

// ---------------------------------------------------------------------------
// Kernel 1: GroupNorm statistics. One block per (b, g).
// ---------------------------------------------------------------------------
__global__ __launch_bounds__(TPB) void gn_stats_kernel(const float* __restrict__ x,
                                                       float* __restrict__ stats) {
  const int bg = blockIdx.x;  // 0..63
  const float* p = x + (size_t)bg * 96 * 4096;
  float s = 0.f, s2 = 0.f;
  for (int i = threadIdx.x; i < 96 * 4096; i += TPB) {
    float v = p[i];
    s += v; s2 += v * v;
  }
  __shared__ float r1[TPB];
  __shared__ float r2[TPB];
  r1[threadIdx.x] = s; r2[threadIdx.x] = s2;
  __syncthreads();
  for (int off = TPB / 2; off > 0; off >>= 1) {
    if (threadIdx.x < off) {
      r1[threadIdx.x] += r1[threadIdx.x + off];
      r2[threadIdx.x] += r2[threadIdx.x + off];
    }
    __syncthreads();
  }
  if (threadIdx.x == 0) {
    const float N = 96.f * 4096.f;
    float mean = r1[0] / N;
    float var  = r2[0] / N - mean * mean;
    stats[bg * 2 + 0] = mean;
    stats[bg * 2 + 1] = rsqrtf(var + 1e-5f);
  }
}

// ---------------------------------------------------------------------------
// Kernel 2: pack row-major W[N][K] (f32) into WMMA-B fragment order (f16):
// packed[(kt*Nt+nt)*512 + lane*16 + j] = W[nt*16+(lane&15)][kt*32+(lane>=16?16:0)+j]
// ---------------------------------------------------------------------------
__global__ __launch_bounds__(TPB) void pack_weight_kernel(const float* __restrict__ W,
                                                          _Float16* __restrict__ packed,
                                                          int K, int Ntiles, int total) {
  int gid = blockIdx.x * TPB + threadIdx.x;
  if (gid >= total) return;
  int tile = gid >> 9;
  int r    = gid & 511;
  int lane = r >> 4;
  int j    = r & 15;
  int kt = tile / Ntiles;
  int nt = tile - kt * Ntiles;
  int n = nt * 16 + (lane & 15);
  int k = kt * 32 + ((lane >> 4) << 4) + j;
  packed[gid] = (_Float16)W[(size_t)n * K + k];
}

// ---------------------------------------------------------------------------
// Kernel 3: fused per-window GN-apply + QKV + attention + proj + residual.
// LDS layout (279,552 B dynamic):
//   ywin/hbuf : [64][392] f16  @0        (window activations; later attn out)
//   qm        : [8][64][64] f16 @50176   (q*scale, hd padded to 64)
//   km        : [8][64][64] f16 @115712
//   vT        : [8][48][64] f16 @181248  ([head][d][s])
//   sf32      : [8][16][64] f32 @230400
//   sf16      : [8][16][64] f16 @263168
//   xwin(f32) : [384][64] f32  @50176    (TDM staging; overlays qm + km head)
// ---------------------------------------------------------------------------
__global__ __launch_bounds__(TPB) void fused_window_attn_kernel(
    const float* __restrict__ x, const float* __restrict__ gnw,
    const float* __restrict__ gnb, const float* __restrict__ stats,
    const _Float16* __restrict__ wqkv_p, const _Float16* __restrict__ wproj_p,
    const float* __restrict__ bproj, float* __restrict__ out) {
  extern __shared__ __align__(16) char smem[];
  _Float16* ywin = (_Float16*)smem;                  // 25088 halves
  _Float16* qm   = ywin + 64 * 392;                  // 32768 halves
  _Float16* km   = qm + 8 * 64 * 64;                 // 32768 halves
  _Float16* vT   = km + 8 * 64 * 64;                 // 24576 halves
  float*    sf32 = (float*)(vT + 8 * 48 * 64);       // 8192 floats
  _Float16* sf16 = (_Float16*)(sf32 + 8 * 16 * 64);  // 8192 halves
  float*    xwin = (float*)qm;                       // staging overlay [c][s]

  const int tid  = threadIdx.x;
  const int lane = tid & 31;
  const int wave = tid >> 5;
  const int wi = blockIdx.x;  // b*64 + wh*8 + ww
  const int b  = wi >> 6;
  const int wh = (wi >> 3) & 7;
  const int ww = wi & 7;
  const size_t xbase = (size_t)b * 384 * 4096;

  // ---- stage 0a: TDM load of the raw x window into LDS ------------------
  // 3-D tile: X=8 (w, contiguous), Y=8 (h, stride 64), Z=384 (c, stride 4096)
  // -> LDS layout xwin[c][h][w] = xwin[c][s], f32, 98,304 B.
  if (tid == 0) {
    uint64_t ga = (uint64_t)(uintptr_t)(x + xbase + (size_t)(wh * 8 * 64 + ww * 8));
    uint32_t lds_off = (uint32_t)(uintptr_t)(void*)xwin;  // LDS byte offset
    uv4 g0;
    g0[0] = 1u;                                   // count=1, user descriptor
    g0[1] = lds_off;                              // lds_addr
    g0[2] = (uint32_t)ga;                         // global_addr[31:0]
    g0[3] = (uint32_t)((ga >> 32) & 0x01ffffffu) | (2u << 30);  // type=2
    uv8 g1;
    g1[0] = 2u << 16;                             // data_size = 4 bytes
    g1[1] = 64u << 16;                            // tensor_dim0 = 64 (low16)
    g1[2] = 64u << 16;                            // dim0 hi=0 | tensor_dim1=64 lo
    g1[3] = 8u << 16;                             // dim1 hi=0 | tile_dim0 = 8
    g1[4] = 8u | (384u << 16);                    // tile_dim1=8, tile_dim2=384
    g1[5] = 64u;                                  // tensor_dim0_stride = 64
    g1[6] = 4096u << 16;                          // tensor_dim1_stride = 4096
    g1[7] = 0u;
    uv4 g2; g2[0] = 384u; g2[1] = 0u; g2[2] = 0u; g2[3] = 0u;  // tensor_dim2
    uv4 g3; g3[0] = 0u; g3[1] = 0u; g3[2] = 0u; g3[3] = 0u;
    asm volatile("tensor_load_to_lds %0, %1, %2, %3"
                 :: "s"(g0), "s"(g1), "s"(g2), "s"(g3) : "memory");
    __builtin_amdgcn_s_wait_tensorcnt(0);
  }
  __syncthreads();

  // ---- stage 0b: GroupNorm affine from LDS staging, f32 -> f16 ----------
  for (int e = tid; e < 64 * 384; e += TPB) {
    int s = e & 63, c = e >> 6;
    float xv = xwin[c * 64 + s];
    int g = c / 96;
    float mean = stats[(b * 4 + g) * 2 + 0];
    float rstd = stats[(b * 4 + g) * 2 + 1];
    ywin[s * 392 + c] = (_Float16)((xv - mean) * rstd * gnw[c] + gnb[c]);
  }
  __syncthreads();   // xwin dead; qm/km region reusable

  // ---- zero the hd padding columns (d = 48..63) of qm and km ------------
  for (int i = tid; i < 2 * 512 * 16; i += TPB) {
    int m = i >> 4;
    qm[(m >> 9) * 32768 + (m & 511) * 64 + 48 + (i & 15)] = (_Float16)0.f;
  }

  // ---- stage 1: QKV GEMM (M=64, N=1152, K=384), software-pipelined ------
  const float scale = 0.14433756729740643f;  // 48^-0.5
  const int laneA = (lane & 15) * 392 + ((lane >> 4) << 4);
  const int laneB = lane * 16;
  for (int t = wave; t < 4 * 72; t += 8) {
    int mt = t / 72, nt = t - mt * 72;
    const _Float16* abase = ywin + mt * 16 * 392 + laneA;
    const _Float16* bbase = wqkv_p + nt * 512 + laneB;
    fv8 acc = {};
    hv16 a = lda_off(abase, 0);
    hv16 bm = ldb_off(bbase, 0);
#pragma unroll
    for (int kt = 0; kt < 12; ++kt) {
      hv16 an = a, bn = bm;
      if (kt < 11) {
        an = lda_off(abase, 32 * (kt + 1));
        bn = ldb_off(bbase, 36864 * (kt + 1));  // 72*512 halves per K tile
      }
      acc = WMMA_F16(a, bm, acc);
      a = an; bm = bn;
    }
    int o0    = nt * 16;                // tile never straddles a head
    int which = o0 / 384;               // 0=q,1=k,2=v
    int head  = (o0 % 384) / 48;
    int d     = (o0 % 48) + (lane & 15);
    int mrow  = (lane >> 4) << 3;
#pragma unroll
    for (int i = 0; i < 8; ++i) {
      int s = mt * 16 + mrow + i;
      float v = acc[i];
      if (which == 0)      qm[(head * 64 + s) * 64 + d] = (_Float16)(v * scale);
      else if (which == 1) km[(head * 64 + s) * 64 + d] = (_Float16)v;
      else                 vT[(head * 48 + d) * 64 + s] = (_Float16)v;
    }
  }
  __syncthreads();

  // ---- stage 2: attention, one head per wave ----------------------------
  {
    const int hd = wave;
    float*    myf32 = sf32 + hd * 16 * 64;
    _Float16* myf16 = sf16 + hd * 16 * 64;
    const _Float16* qh = qm + hd * 64 * 64;
    const _Float16* kh = km + hd * 64 * 64;
    const _Float16* vh = vT + hd * 48 * 64;
    const int laneA64 = (lane & 15) * 64 + ((lane >> 4) << 4);
    const int laneB64 = (lane & 15) * 64 + ((lane >> 4) << 4);
    const _Float16* kl = kh + laneB64;
    const _Float16* vl = vh + laneB64;
    const _Float16* al = myf16 + laneA64;

    for (int mt = 0; mt < 4; ++mt) {
      const _Float16* ql = qh + mt * 1024 + laneA64;
      fv8 sc0 = {}, sc1 = {}, sc2 = {}, sc3 = {};
#pragma unroll
      for (int kk = 0; kk < 2; ++kk) {
        hv16 a  = lda_off(ql, kk * 32);
        hv16 b0 = ldb_off(kl, 0 * 1024 + kk * 32);
        hv16 b1 = ldb_off(kl, 1 * 1024 + kk * 32);
        hv16 b2 = ldb_off(kl, 2 * 1024 + kk * 32);
        hv16 b3 = ldb_off(kl, 3 * 1024 + kk * 32);
        sc0 = WMMA_F16(a, b0, sc0);
        sc1 = WMMA_F16(a, b1, sc1);
        sc2 = WMMA_F16(a, b2, sc2);
        sc3 = WMMA_F16(a, b3, sc3);
      }
      int nl = lane & 15, mb = (lane >> 4) << 3;
#pragma unroll
      for (int i = 0; i < 8; ++i) {
        myf32[(mb + i) * 64 +  0 + nl] = sc0[i];
        myf32[(mb + i) * 64 + 16 + nl] = sc1[i];
        myf32[(mb + i) * 64 + 32 + nl] = sc2[i];
        myf32[(mb + i) * 64 + 48 + nl] = sc3[i];
      }
      // softmax: lanes 0..15 own one query row each (same-wave LDS in-order)
      if (lane < 16) {
        float* row = myf32 + lane * 64;
        float mx = -1e30f;
        for (int n = 0; n < 64; ++n) mx = fmaxf(mx, row[n]);
        float sum = 0.f;
        for (int n = 0; n < 64; ++n) { float e = __expf(row[n] - mx); sum += e; row[n] = e; }
        float inv = 1.f / sum;
        _Float16* row16 = myf16 + lane * 64;
        for (int n = 0; n < 64; ++n) row16[n] = (_Float16)(row[n] * inv);
      }
      // h = attn(16x64) * v(64x48)
#pragma unroll
      for (int nt = 0; nt < 3; ++nt) {
        fv8 acc = {};
#pragma unroll
        for (int kk = 0; kk < 2; ++kk) {
          hv16 a  = lda_off(al, kk * 32);
          hv16 bm = ldb_off(vl, nt * 1024 + kk * 32);  // [n=d][k=s]
          acc = WMMA_F16(a, bm, acc);
        }
        int c = hd * 48 + nt * 16 + (lane & 15);
#pragma unroll
        for (int i = 0; i < 8; ++i) {
          int s = mt * 16 + ((lane >> 4) << 3) + i;
          ywin[s * 392 + c] = (_Float16)acc[i];  // hbuf reuses ywin region
        }
      }
    }
  }
  __syncthreads();

  // ---- stage 3: output proj (M=64, N=384, K=384) + bias + residual ------
  const _Float16* hbuf = ywin;
  for (int t = wave; t < 4 * 24; t += 8) {
    int mt = t / 24, nt = t - mt * 24;
    const _Float16* abase = hbuf + mt * 16 * 392 + laneA;
    const _Float16* bbase = wproj_p + nt * 512 + laneB;
    fv8 acc = {};
    hv16 a = lda_off(abase, 0);
    hv16 bm = ldb_off(bbase, 0);
#pragma unroll
    for (int kt = 0; kt < 12; ++kt) {
      hv16 an = a, bn = bm;
      if (kt < 11) {
        an = lda_off(abase, 32 * (kt + 1));
        bn = ldb_off(bbase, 12288 * (kt + 1));  // 24*512 halves per K tile
      }
      acc = WMMA_F16(a, bm, acc);
      a = an; bm = bn;
    }
    int co = nt * 16 + (lane & 15);
    float bp = bproj[co];
#pragma unroll
    for (int i = 0; i < 8; ++i) {
      int s  = mt * 16 + ((lane >> 4) << 3) + i;
      int hh = wh * 8 + (s >> 3), wp = ww * 8 + (s & 7);
      size_t oi = xbase + (size_t)co * 4096 + hh * 64 + wp;
      out[oi] = acc[i] + bp + x[oi];
    }
  }
}

// ---------------------------------------------------------------------------
// Host entry
// ---------------------------------------------------------------------------
extern "C" void kernel_launch(void* const* d_in, const int* in_sizes, int n_in,
                              void* d_out, int out_size, void* d_ws, size_t ws_size,
                              hipStream_t stream) {
  const float* x     = (const float*)d_in[0];
  const float* gnw   = (const float*)d_in[1];
  const float* gnb   = (const float*)d_in[2];
  const float* wqkv  = (const float*)d_in[3];  // [1152][384]
  const float* wproj = (const float*)d_in[4];  // [384][384]
  const float* bproj = (const float*)d_in[5];
  float* out = (float*)d_out;

  char* ws = (char*)d_ws;
  float*    stats   = (float*)ws;                                   // 64*2 f32
  _Float16* wqkv_p  = (_Float16*)(ws + 1024);                       // 12*72*512 h
  _Float16* wproj_p = (_Float16*)(ws + 1024 + (size_t)12*72*512*2); // 12*24*512 h

  const int qkv_total  = 12 * 72 * 512;
  const int proj_total = 12 * 24 * 512;

  gn_stats_kernel<<<64, TPB, 0, stream>>>(x, stats);
  pack_weight_kernel<<<(qkv_total + TPB - 1) / TPB, TPB, 0, stream>>>(
      wqkv, wqkv_p, 384, 72, qkv_total);
  pack_weight_kernel<<<(proj_total + TPB - 1) / TPB, TPB, 0, stream>>>(
      wproj, wproj_p, 384, 24, proj_total);

  const size_t LDS_BYTES = (size_t)(64 * 392 + 2 * 8 * 64 * 64 + 8 * 48 * 64) * 2
                         + (size_t)8 * 16 * 64 * 4
                         + (size_t)8 * 16 * 64 * 2;  // 279,552 B
  (void)hipFuncSetAttribute((const void*)fused_window_attn_kernel,
                            hipFuncAttributeMaxDynamicSharedMemorySize,
                            (int)LDS_BYTES);
  fused_window_attn_kernel<<<1024, TPB, LDS_BYTES, stream>>>(
      x, gnw, gnb, stats, wqkv_p, wproj_p, bproj, out);
}
